// BERT_BiLSTM_CRF_14774687498665
// MI455X (gfx1250) — matrix operand order repaired and checked by reading, not compiled, not run on hardware
//
#include <hip/hip_runtime.h>
#include <hip/hip_bf16.h>

// ---------------------------------------------------------------------------
// BERT-base (B=4,S=256,H=768,L=12,NH=12,DH=64,FF=3072) + 2x BiLSTM(R=128)
// + CRF(T=9) forward NLL on gfx1250.
//
// All dense math routed through V_WMMA_F32_16X16X32_BF16 (fp32 accumulate,
// bf16 inputs converted in LDS). Model is bandwidth bound (~0.44 GB weights,
// ~0.2 TFLOP at 23.3 TB/s HBM / 192 MB L2), so epilogues (bias / exact GELU /
// attention scale+mask) are fused into the GEMM and activations stay fp32.
//
// GEMM tile: 128(M) x 64(N) x 32(K) per 256-thread (8-wave) workgroup.
// Each wave owns a 16x64 strip = 4 WMMA accumulators, so one A-fragment
// feeds 4 v_wmma issues per K-step (4x math per LDS byte vs 1-accum form).
// All M in this net are multiples of 128 -> no M bounds checks anywhere.
// ---------------------------------------------------------------------------

typedef __attribute__((ext_vector_type(16))) __bf16 v16bf;
typedef __attribute__((ext_vector_type(8)))  float  v8f;

union Frag { v16bf v; unsigned int u[8]; };

#define FLAG_BIAS 1
#define FLAG_GELU 2
#define FLAG_MASK 4

__device__ __forceinline__ unsigned int pack_bf2(float a, float b) {
  unsigned int ua = __float_as_uint(a), ub = __float_as_uint(b);
  ua = (ua + 0x7FFFu + ((ua >> 16) & 1u)) >> 16;  // RNE fp32 -> bf16
  ub = (ub + 0x7FFFu + ((ub >> 16) & 1u)) >> 16;
  return (ua & 0xFFFFu) | (ub << 16);
}

// Generic batched GEMM: C[z] = epilogue(A[z] x B[z]).
// A elem (m,k) = A[zo*aStrO + zi*aStrI + m*lda + k]      (k contiguous)
// B elem (k,n) = B[zo*bStrO + zi*bStrI + k*ldbK + n*ldbN]
// C elem (m,n) = C[zo*cStrO + zi*cStrI + m*ldc + n]
// zo = z/zdiv, zi = z%zdiv  (one launch covers [B,NH] batched attention).
// REQUIRES: M % 128 == 0, K % 32 == 0. N may be ragged (bounds-checked).
__global__ __launch_bounds__(256) void gemm_bf16_wmma(
    const float* __restrict__ A, long aStrO, long aStrI, int lda,
    const float* __restrict__ Bm, long bStrO, long bStrI,
    long ldbK, int ldbN,
    float* __restrict__ C, long cStrO, long cStrI, int ldc,
    int M, int N, int K, int zdiv,
    const float* __restrict__ bias, const int* __restrict__ mask,
    float scale, int flags)
{
  __shared__ unsigned int lds_a[128 * 16];  // 128 rows x 32 bf16 (16 dwords)
  __shared__ unsigned int lds_b[64 * 17];   // 64 rows x 32 bf16 (+pad dword)

  const int tid = threadIdx.x;
  const int z  = blockIdx.z;
  const int zo = z / zdiv, zi = z % zdiv;
  const float* Ab = A  + (long)zo * aStrO + (long)zi * aStrI;
  const float* Bb = Bm + (long)zo * bStrO + (long)zi * bStrI;
  float*       Cb = C  + (long)zo * cStrO + (long)zi * cStrI;

  const int mBase = blockIdx.y * 128;
  const int nBase = blockIdx.x * 64;
  const int wave = tid >> 5, lane = tid & 31;
  // wave -> 16-row strip of the 128-row tile; covers all 64 columns.
  const int waveM = mBase + (wave << 4);

  v8f acc[4] = {};

  for (int k0 = 0; k0 < K; k0 += 32) {
    __syncthreads();
    // Stage A tile (128x32) as bf16 pairs, K-contiguous. M always in range.
#pragma unroll
    for (int it = 0; it < 8; ++it) {
      int p = tid + it * 256;               // 0..2047 pair index
      int row = p >> 4, kp = (p & 15) << 1;
      const float* src = Ab + (long)(mBase + row) * lda + (k0 + kp);
      lds_a[p] = pack_bf2(src[0], src[1]);
    }
    // Stage B tile (32x64) transposed into LDS as [n][k] (K-contiguous).
#pragma unroll
    for (int it = 0; it < 4; ++it) {
      int p = tid + it * 256;               // 0..1023 pair index
      int nIdx = p >> 4, kp = (p & 15) << 1;
      int n = nBase + nIdx;
      float f0 = 0.f, f1 = 0.f;
      if (n < N) {
        long base = (long)(k0 + kp) * ldbK + (long)n * ldbN;
        f0 = Bb[base]; f1 = Bb[base + ldbK];
      }
      lds_b[nIdx * 17 + (p & 15)] = pack_bf2(f0, f1);
    }
    __syncthreads();

    if (k0 + 32 < K)  // hint next K-tile toward GL2 (global_prefetch)
      __builtin_prefetch(Ab + (long)(waveM + (lane & 15)) * lda + (k0 + 32), 0, 1);

    // A fragment: lanes 0-15 carry K 0-7/16-23, lanes 16-31 carry 8-15/24-31.
    Frag fa;
    const int rowA = (wave << 4) + (lane & 15);
    const int kg = (lane >> 4) << 3;
#pragma unroll
    for (int d = 0; d < 8; ++d) {
      int kk = kg + ((d & 4) ? 16 : 0) + ((d & 3) << 1);
      fa.u[d] = lds_a[rowA * 16 + (kk >> 1)];
    }
    // 4 B fragments: lanes 0-15 K 0-7/16-23, lanes 16-31 K 8-15/24-31, lane%16 = n.
    const int kgB = (lane >> 4) << 2;       // dword base: 0 or 4
#pragma unroll
    for (int t = 0; t < 4; ++t) {
      Frag fb;
      const int rowB = (t << 4) + (lane & 15);
#pragma unroll
      for (int d = 0; d < 8; ++d)
        fb.u[d] = lds_b[rowB * 17 + kgB + (d & 3) + ((d & 4) ? 8 : 0)];
      acc[t] = __builtin_amdgcn_wmma_f32_16x16x32_bf16(
          false, fa.v, false, fb.v, (short)0, acc[t], false, false);
    }
  }

  // Epilogue. C layout: VGPR r -> M = r + 8*(lane/16), N = lane%16.
  const int mRow = waveM + ((lane >> 4) << 3);
#pragma unroll
  for (int t = 0; t < 4; ++t) {
    int n = nBase + (t << 4) + (lane & 15);
    if (n >= N) continue;
    float maskAdd = 0.f, bv = 0.f;
    if (flags & FLAG_MASK) maskAdd = (1.f - (float)mask[zo * N + n]) * -1e9f;
    if (flags & FLAG_BIAS) bv = bias[n];
#pragma unroll
    for (int r = 0; r < 8; ++r) {
      float v = acc[t][r];
      if (flags & FLAG_MASK) v = v * scale + maskAdd;
      v += bv;
      if (flags & FLAG_GELU) v = 0.5f * v * (1.f + erff(v * 0.70710678118f));
      Cb[(long)(mRow + r) * ldc + n] = v;
    }
  }
}

// out[m,:] = word_emb[ids[m]] + pos_emb[m%S] + type_emb[0]
__global__ __launch_bounds__(256) void embed_kernel(
    const int* __restrict__ ids, const float* __restrict__ we,
    const float* __restrict__ pe, const float* __restrict__ te,
    float* __restrict__ out)
{
  int m = blockIdx.x, s = m & 255;
  long id = ids[m];
  for (int h = threadIdx.x; h < 768; h += 256)
    out[(long)m * 768 + h] = we[id * 768 + h] + pe[(long)s * 768 + h] + te[h];
}

// out = LN(src [+ res]) * g + b   over rows of length 768, eps = 1e-12
__global__ __launch_bounds__(256) void ln_kernel(
    const float* __restrict__ src, const float* __restrict__ res,
    const float* __restrict__ g, const float* __restrict__ b,
    float* __restrict__ out)
{
  __shared__ float red[256];
  const long row = blockIdx.x;
  const float* s = src + row * 768;
  const float* r = res ? res + row * 768 : nullptr;
  float v[3], sum = 0.f;
#pragma unroll
  for (int i = 0; i < 3; ++i) {
    int h = threadIdx.x + i * 256;
    float x = s[h];
    if (r) x += r[h];
    v[i] = x; sum += x;
  }
  red[threadIdx.x] = sum; __syncthreads();
  for (int o = 128; o > 0; o >>= 1) {
    if (threadIdx.x < o) red[threadIdx.x] += red[threadIdx.x + o];
    __syncthreads();
  }
  float mu = red[0] * (1.f / 768.f);
  __syncthreads();
  float sq = 0.f;
#pragma unroll
  for (int i = 0; i < 3; ++i) { float d = v[i] - mu; sq += d * d; }
  red[threadIdx.x] = sq; __syncthreads();
  for (int o = 128; o > 0; o >>= 1) {
    if (threadIdx.x < o) red[threadIdx.x] += red[threadIdx.x + o];
    __syncthreads();
  }
  float inv = rsqrtf(red[0] * (1.f / 768.f) + 1e-12f);
#pragma unroll
  for (int i = 0; i < 3; ++i) {
    int h = threadIdx.x + i * 256;
    out[row * 768 + h] = (v[i] - mu) * inv * g[h] + b[h];
  }
}

// In-place softmax over rows of length 256 (one block per row).
__global__ __launch_bounds__(256) void softmax_kernel(float* __restrict__ x) {
  __shared__ float red[256];
  float* p = x + (long)blockIdx.x * 256;
  float v = p[threadIdx.x];
  red[threadIdx.x] = v; __syncthreads();
  for (int o = 128; o > 0; o >>= 1) {
    if (threadIdx.x < o) red[threadIdx.x] = fmaxf(red[threadIdx.x], red[threadIdx.x + o]);
    __syncthreads();
  }
  float mx = red[0]; __syncthreads();
  float e = expf(v - mx);
  red[threadIdx.x] = e; __syncthreads();
  for (int o = 128; o > 0; o >>= 1) {
    if (threadIdx.x < o) red[threadIdx.x] += red[threadIdx.x + o];
    __syncthreads();
  }
  p[threadIdx.x] = e / red[0];
}

__global__ void addvec_kernel(const float* a, const float* b, float* o, int n) {
  int i = blockIdx.x * 256 + threadIdx.x;
  if (i < n) o[i] = a[i] + b[i];
}

// Persistent per-(batch) LSTM direction scan. xg = x@Wih^T + bih + bhh
// precomputed. Gates i,f,g,o (PyTorch order). Writes out[b,t,dirOff+j].
__global__ __launch_bounds__(256) void lstm_scan_kernel(
    const float* __restrict__ xg, const float* __restrict__ Whh,
    float* __restrict__ out, int dirOff, int reverse)
{
  __shared__ float h[128], c[128], g[512];
  const int b = blockIdx.x, tid = threadIdx.x;
  if (tid < 128) { h[tid] = 0.f; c[tid] = 0.f; }
  __syncthreads();
  for (int s = 0; s < 256; ++s) {
    const int t = reverse ? 255 - s : s;
    const float* xr = xg + ((long)b * 256 + t) * 512;
    for (int j = tid; j < 512; j += 256) {
      float acc = xr[j];
      const float* w = Whh + (long)j * 128;
#pragma unroll 4
      for (int k = 0; k < 128; ++k) acc += h[k] * w[k];
      g[j] = acc;
    }
    __syncthreads();
    if (tid < 128) {
      float si = 1.f / (1.f + expf(-g[tid]));
      float sf = 1.f / (1.f + expf(-g[128 + tid]));
      float gg = tanhf(g[256 + tid]);
      float so = 1.f / (1.f + expf(-g[384 + tid]));
      float cc = sf * c[tid] + si * gg;
      c[tid] = cc;
      float hh = so * tanhf(cc);
      h[tid] = hh;
      out[((long)b * 256 + t) * 256 + dirOff + tid] = hh;
    }
    __syncthreads();
  }
}

// CRF negative log-likelihood (sum over batch). em:[4,256,9].
__global__ __launch_bounds__(64) void crf_kernel(
    const float* __restrict__ em, const int* __restrict__ tags,
    const int* __restrict__ mask, const float* __restrict__ start,
    const float* __restrict__ endv, const float* __restrict__ trans,
    float* __restrict__ out)
{
  __shared__ float sc[2][4 * 16];
  __shared__ float numv[4], denv[4];
  const int tid = threadIdx.x;
  const int b = tid >> 4, j = tid & 15;
  if (j < 9) sc[0][b * 16 + j] = start[j] + em[((long)b * 256) * 9 + j];
  __syncthreads();
  int cur = 0;
  for (int s = 1; s < 256; ++s) {
    float nv = 0.f;
    if (j < 9) {
      float mx = -1e30f;
      for (int i = 0; i < 9; ++i)
        mx = fmaxf(mx, sc[cur][b * 16 + i] + trans[i * 9 + j]);
      float ss = 0.f;
      for (int i = 0; i < 9; ++i)
        ss += expf(sc[cur][b * 16 + i] + trans[i * 9 + j] - mx);
      float nxt = mx + logf(ss) + em[((long)b * 256 + s) * 9 + j];
      nv = (mask[b * 256 + s] > 0) ? nxt : sc[cur][b * 16 + j];
    }
    __syncthreads();
    if (j < 9) sc[cur ^ 1][b * 16 + j] = nv;
    __syncthreads();
    cur ^= 1;
  }
  if (tid < 4) {
    float mx = -1e30f;
    for (int t = 0; t < 9; ++t) mx = fmaxf(mx, sc[cur][tid * 16 + t] + endv[t]);
    float ss = 0.f;
    for (int t = 0; t < 9; ++t) ss += expf(sc[cur][tid * 16 + t] + endv[t] - mx);
    denv[tid] = mx + logf(ss);
    int t0 = tags[tid * 256];
    float num = start[t0] + em[((long)tid * 256) * 9 + t0];
    int last = -1;
    for (int s = 0; s < 256; ++s) last += (mask[tid * 256 + s] > 0) ? 1 : 0;
    for (int s = 1; s < 256; ++s) {
      if (mask[tid * 256 + s] > 0) {
        int tp = tags[tid * 256 + s - 1], tc = tags[tid * 256 + s];
        num += trans[tp * 9 + tc] + em[((long)tid * 256 + s) * 9 + tc];
      }
    }
    num += endv[tags[tid * 256 + last]];
    numv[tid] = num;
  }
  __syncthreads();
  if (tid == 0) {
    float tot = 0.f;
    for (int bb = 0; bb < 4; ++bb) tot += denv[bb] - numv[bb];
    out[0] = tot;
  }
}

extern "C" void kernel_launch(void* const* d_in, const int* in_sizes, int n_in,
                              void* d_out, int out_size, void* d_ws, size_t ws_size,
                              hipStream_t stream) {
  (void)in_sizes; (void)n_in; (void)out_size; (void)ws_size;
  // setup_inputs() insertion order, params flattened recursively.
  const int* input_ids = (const int*)d_in[0];
  const int* attn_mask = (const int*)d_in[1];
  const int* tags      = (const int*)d_in[2];
  const float* word_emb = (const float*)d_in[3];
  const float* pos_emb  = (const float*)d_in[4];
  const float* type_emb = (const float*)d_in[5];
  const float* emb_ln_g = (const float*)d_in[6];
  const float* emb_ln_b = (const float*)d_in[7];
  const float* Wq = (const float*)d_in[8];   const float* bqv = (const float*)d_in[9];
  const float* Wk = (const float*)d_in[10];  const float* bkv = (const float*)d_in[11];
  const float* Wv = (const float*)d_in[12];  const float* bvv = (const float*)d_in[13];
  const float* Wo = (const float*)d_in[14];  const float* bov = (const float*)d_in[15];
  const float* ln1g = (const float*)d_in[16]; const float* ln1b = (const float*)d_in[17];
  const float* W1 = (const float*)d_in[18];  const float* b1v = (const float*)d_in[19];
  const float* W2 = (const float*)d_in[20];  const float* b2v = (const float*)d_in[21];
  const float* ln2g = (const float*)d_in[22]; const float* ln2b = (const float*)d_in[23];
  const float* l0Wihf = (const float*)d_in[24]; const float* l0Whhf = (const float*)d_in[25];
  const float* l0bihf = (const float*)d_in[26]; const float* l0bhhf = (const float*)d_in[27];
  const float* l0Wihb = (const float*)d_in[28]; const float* l0Whhb = (const float*)d_in[29];
  const float* l0bihb = (const float*)d_in[30]; const float* l0bhhb = (const float*)d_in[31];
  const float* l1Wihf = (const float*)d_in[32]; const float* l1Whhf = (const float*)d_in[33];
  const float* l1bihf = (const float*)d_in[34]; const float* l1bhhf = (const float*)d_in[35];
  const float* l1Wihb = (const float*)d_in[36]; const float* l1Whhb = (const float*)d_in[37];
  const float* l1bihb = (const float*)d_in[38]; const float* l1bhhb = (const float*)d_in[39];
  const float* cls_W = (const float*)d_in[40]; const float* cls_b = (const float*)d_in[41];
  const float* crf_start = (const float*)d_in[42];
  const float* crf_end   = (const float*)d_in[43];
  const float* crf_trans = (const float*)d_in[44];

  const int Mn = 1024, Hh = 768, FFc = 3072, Sc = 256;

  float* ws = (float*)d_ws;
  float* bufX  = ws;                 ws += (long)Mn * Hh;        // 786432
  float* bufY  = ws;                 ws += (long)Mn * Hh;
  float* qb    = ws;                 ws += (long)Mn * Hh;
  float* kb    = ws;                 ws += (long)Mn * Hh;
  float* vb    = ws;                 ws += (long)Mn * Hh;
  float* scb   = ws;                 ws += (long)48 * Sc * Sc;   // 3145728
  float* ctxb  = ws;                 ws += (long)Mn * Hh;
  float* ffb   = ws;                 ws += (long)Mn * FFc;       // 3145728
  float* tmpb  = ws;                 ws += (long)Mn * Hh;
  float* xgf   = ws;                 ws += (long)Mn * 512;
  float* xgb   = ws;                 ws += (long)Mn * 512;
  float* lx    = ws;                 ws += (long)Mn * 256;
  float* ly    = ws;                 ws += (long)Mn * 256;
  float* biasf = ws;                 ws += 512;
  float* biasb = ws;                 ws += 512;
  float* emb   = ws;                 ws += (long)Mn * 9;

  auto gemm = [&](const float* A, long aSO, long aSI, int lda,
                  const float* Bm, long bSO, long bSI, long ldbK, int ldbN,
                  float* C, long cSO, long cSI, int ldc,
                  int M, int N, int K, int batch, int zdiv,
                  const float* bias, const int* mask, float scale, int flags) {
    dim3 grid((N + 63) / 64, M / 128, batch);
    gemm_bf16_wmma<<<grid, dim3(256), 0, stream>>>(
        A, aSO, aSI, lda, Bm, bSO, bSI, ldbK, ldbN,
        C, cSO, cSI, ldc, M, N, K, zdiv, bias, mask, scale, flags);
  };

  // ---- embeddings + LN ----
  embed_kernel<<<Mn, 256, 0, stream>>>(input_ids, word_emb, pos_emb, type_emb, tmpb);
  ln_kernel<<<Mn, 256, 0, stream>>>(tmpb, nullptr, emb_ln_g, emb_ln_b, bufX);

  // ---- 12 encoder layers ----
  for (int l = 0; l < 12; ++l) {
    const long wHH = (long)l * Hh * Hh, vH = (long)l * Hh;
    // QKV projections (x @ W + b)
    gemm(bufX, 0, 0, Hh, Wq + wHH, 0, 0, Hh, 1, qb, 0, 0, Hh,
         Mn, Hh, Hh, 1, 1, bqv + vH, nullptr, 1.f, FLAG_BIAS);
    gemm(bufX, 0, 0, Hh, Wk + wHH, 0, 0, Hh, 1, kb, 0, 0, Hh,
         Mn, Hh, Hh, 1, 1, bkv + vH, nullptr, 1.f, FLAG_BIAS);
    gemm(bufX, 0, 0, Hh, Wv + wHH, 0, 0, Hh, 1, vb, 0, 0, Hh,
         Mn, Hh, Hh, 1, 1, bvv + vH, nullptr, 1.f, FLAG_BIAS);
    // scores[b,h] = (q . k^T) / 8 + mask_bias   (48-way batched, zdiv=NH)
    gemm(qb, (long)Sc * Hh, 64, Hh,
         kb, (long)Sc * Hh, 64, 1, Hh,
         scb, (long)12 * Sc * Sc, (long)Sc * Sc, Sc,
         Sc, Sc, 64, 48, 12, nullptr, attn_mask, 0.125f, FLAG_MASK);
    softmax_kernel<<<48 * Sc, 256, 0, stream>>>(scb);
    // ctx[b,h] = probs . v
    gemm(scb, (long)12 * Sc * Sc, (long)Sc * Sc, Sc,
         vb, (long)Sc * Hh, 64, Hh, 1,
         ctxb, (long)Sc * Hh, 64, Hh,
         Sc, 64, Sc, 48, 12, nullptr, nullptr, 1.f, 0);
    // output projection + residual LN
    gemm(ctxb, 0, 0, Hh, Wo + wHH, 0, 0, Hh, 1, tmpb, 0, 0, Hh,
         Mn, Hh, Hh, 1, 1, bov + vH, nullptr, 1.f, FLAG_BIAS);
    ln_kernel<<<Mn, 256, 0, stream>>>(tmpb, bufX, ln1g + vH, ln1b + vH, bufY);
    // FFN: GELU(x@W1+b1) @ W2 + b2, residual LN
    gemm(bufY, 0, 0, Hh, W1 + (long)l * Hh * FFc, 0, 0, FFc, 1, ffb, 0, 0, FFc,
         Mn, FFc, Hh, 1, 1, b1v + (long)l * FFc, nullptr, 1.f, FLAG_BIAS | FLAG_GELU);
    gemm(ffb, 0, 0, FFc, W2 + (long)l * FFc * Hh, 0, 0, Hh, 1, tmpb, 0, 0, Hh,
         Mn, Hh, FFc, 1, 1, b2v + vH, nullptr, 1.f, FLAG_BIAS);
    ln_kernel<<<Mn, 256, 0, stream>>>(tmpb, bufY, ln2g + vH, ln2b + vH, bufX);
  }

  // ---- BiLSTM layer 0 (input 768) ----
  addvec_kernel<<<2, 256, 0, stream>>>(l0bihf, l0bhhf, biasf, 512);
  addvec_kernel<<<2, 256, 0, stream>>>(l0bihb, l0bhhb, biasb, 512);
  gemm(bufX, 0, 0, Hh, l0Wihf, 0, 0, 1, Hh, xgf, 0, 0, 512,
       Mn, 512, Hh, 1, 1, biasf, nullptr, 1.f, FLAG_BIAS);
  gemm(bufX, 0, 0, Hh, l0Wihb, 0, 0, 1, Hh, xgb, 0, 0, 512,
       Mn, 512, Hh, 1, 1, biasb, nullptr, 1.f, FLAG_BIAS);
  lstm_scan_kernel<<<4, 256, 0, stream>>>(xgf, l0Whhf, lx, 0, 0);
  lstm_scan_kernel<<<4, 256, 0, stream>>>(xgb, l0Whhb, lx, 128, 1);

  // ---- BiLSTM layer 1 (input 256) ----
  addvec_kernel<<<2, 256, 0, stream>>>(l1bihf, l1bhhf, biasf, 512);
  addvec_kernel<<<2, 256, 0, stream>>>(l1bihb, l1bhhb, biasb, 512);
  gemm(lx, 0, 0, 256, l1Wihf, 0, 0, 1, 256, xgf, 0, 0, 512,
       Mn, 512, 256, 1, 1, biasf, nullptr, 1.f, FLAG_BIAS);
  gemm(lx, 0, 0, 256, l1Wihb, 0, 0, 1, 256, xgb, 0, 0, 512,
       Mn, 512, 256, 1, 1, biasb, nullptr, 1.f, FLAG_BIAS);
  lstm_scan_kernel<<<4, 256, 0, stream>>>(xgf, l1Whhf, ly, 0, 0);
  lstm_scan_kernel<<<4, 256, 0, stream>>>(xgb, l1Whhb, ly, 128, 1);

  // ---- classifier + CRF ----
  gemm(ly, 0, 0, 256, cls_W, 0, 0, 9, 1, emb, 0, 0, 9,
       Mn, 9, 256, 1, 1, cls_b, nullptr, 1.f, FLAG_BIAS);
  crf_kernel<<<1, 64, 0, stream>>>(emb, tags, attn_mask, crf_start, crf_end,
                                   crf_trans, (float*)d_out);
}
